// ComplexDFT256_31258771980926
// MI455X (gfx1250) — matrix operand order repaired and checked by reading, not compiled, not run on hardware
//
#include <hip/hip_runtime.h>

typedef __attribute__((ext_vector_type(16))) _Float16 v16h;
typedef __attribute__((ext_vector_type(8)))  float    v8f;

#if defined(__gfx1250__) && \
    __has_builtin(__builtin_amdgcn_global_load_async_to_lds_b128) && \
    __has_builtin(__builtin_amdgcn_s_wait_asynccnt)
#define USE_ASYNC_LDS 1
// The builtin's params are raw int vectors (vector_size(16)) in AS1/AS3.
typedef int v4i __attribute__((vector_size(16)));
typedef __attribute__((address_space(1))) v4i as1_v4i;
typedef __attribute__((address_space(3))) v4i as3_v4i;
#else
#define USE_ASYNC_LDS 0
#endif

// ---------------------------------------------------------------------------
// Kernel 1: build W^T in f16, tiled as [kc=16][n=512][k'=32] (f16), 512 KB.
//   out(B x 512) = X(B x 512) @ W(512 x 512),  W = [[C^T, S^T], [-S^T, C^T]]
//   Wt[n][f] = W[f][n]:
//     n<256,  f<256 :  C[n*256+f]
//     n<256,  f>=256: -S[n*256+(f-256)]
//     n>=256, f<256 :  S[(n-256)*256+f]
//     n>=256, f>=256:  C[(n-256)*256+(f-256)]
// ---------------------------------------------------------------------------
__global__ __launch_bounds__(256) void build_wt_kernel(
    const float* __restrict__ cosk, const float* __restrict__ sink,
    _Float16* __restrict__ wt)
{
    int o   = blockIdx.x * 256 + threadIdx.x;   // 0 .. 262143
    int kc  = o >> 14;                          // / (512*32)
    int rem = o & 16383;
    int n   = rem >> 5;
    int kk  = rem & 31;
    int f   = kc * 32 + kk;

    float v;
    if (n < 256) {
        v = (f < 256) ?  cosk[n * 256 + f] : -sink[n * 256 + (f - 256)];
    } else {
        v = (f < 256) ?  sink[(n - 256) * 256 + f]
                      :  cosk[(n - 256) * 256 + (f - 256)];
    }
    wt[o] = (_Float16)v;
}

// ---------------------------------------------------------------------------
// Kernel 2: out(rows x 512) = X(rows x 512, f32) @ W(512x512, f16), f32 accum.
// Block: 256 threads = 8 waves; block tile 32 rows x 512 cols.
// Wave w: M-subtile (w&1) x 128-col N-group (w>>1) = 8 WMMA tiles, 64 acc VGPRs.
// W slabs (32K x 512N = 32 KB) double-buffered in LDS via async global->LDS.
// ---------------------------------------------------------------------------
__global__ __launch_bounds__(256) void dft_gemm_kernel(
    const float* __restrict__ x, const _Float16* __restrict__ wt,
    float* __restrict__ out)
{
    // 2 buffers x 512 rows x 80B stride (64B data + 16B anti-bank-conflict pad)
    __shared__ uint4 shB[2 * 512 * 5];          // 80 KB

    const int tid  = threadIdx.x;
    const int wv   = tid >> 5;
    const int lane = tid & 31;
    const int hi   = lane >> 4;                 // half-wave select
    const int lcol = lane & 15;
    const int m_sub   = wv & 1;                 // 16-row M tile within block
    const int n_group = wv >> 1;                // 128-col N group

    const int row0 = blockIdx.x * 32 + m_sub * 16 + lcol;   // A row (ISA A layout)
    const float* __restrict__ xrow = x + (size_t)row0 * 512;
    const int koff = hi ? 8 : 0;                // per-lane K split in a 32-chunk

    v8f acc[8];
    #pragma unroll
    for (int t = 0; t < 8; ++t) acc[t] = (v8f){};

    const uint4* __restrict__ wt_u4 = (const uint4*)wt;

    // ---- consume one staged 32-deep K slab from LDS buffer `buf` ------------
    auto compute = [&](int kc, int buf) {
        const int kbase = kc * 32;
        // A fragment: lanes 0-15 K={0..7,16..23}, lanes 16-31 K={8..15,24..31}
        float4 f0 = *(const float4*)(xrow + kbase + koff);
        float4 f1 = *(const float4*)(xrow + kbase + koff + 4);
        float4 f2 = *(const float4*)(xrow + kbase + koff + 16);
        float4 f3 = *(const float4*)(xrow + kbase + koff + 20);
        v16h a;
        a[0]  = (_Float16)f0.x; a[1]  = (_Float16)f0.y;
        a[2]  = (_Float16)f0.z; a[3]  = (_Float16)f0.w;
        a[4]  = (_Float16)f1.x; a[5]  = (_Float16)f1.y;
        a[6]  = (_Float16)f1.z; a[7]  = (_Float16)f1.w;
        a[8]  = (_Float16)f2.x; a[9]  = (_Float16)f2.y;
        a[10] = (_Float16)f2.z; a[11] = (_Float16)f2.w;
        a[12] = (_Float16)f3.x; a[13] = (_Float16)f3.y;
        a[14] = (_Float16)f3.z; a[15] = (_Float16)f3.w;

        const int base = buf * 2560;
        #pragma unroll
        for (int t = 0; t < 8; ++t) {
            const int n = n_group * 128 + t * 16 + lcol;
            union { uint4 u[2]; v16h h; } bb;
            const uint4* p = &shB[base + n * 5 + hi];
            bb.u[0] = p[0];
            bb.u[1] = p[2];
            acc[t] = __builtin_amdgcn_wmma_f32_16x16x32_f16(
                false, a, false, bb.h, (short)0, acc[t], false, false);
        }
    };

#if USE_ASYNC_LDS
    // ---- async stage: 8 x b128 per thread (128B), memory -> LDS, no VGPRs ---
    auto issue = [&](int kc, int buf) {
        const uint4* src = wt_u4 + (size_t)kc * 2048 + tid * 8;
        uint4* dst0 = &shB[buf * 2560 + (tid * 2) * 5];
        uint4* dst1 = dst0 + 5;
        #pragma unroll
        for (int i = 0; i < 4; ++i)
            __builtin_amdgcn_global_load_async_to_lds_b128(
                (as1_v4i*)(src + i), (as3_v4i*)(dst0 + i), 0, 0);
        #pragma unroll
        for (int i = 0; i < 4; ++i)
            __builtin_amdgcn_global_load_async_to_lds_b128(
                (as1_v4i*)(src + 4 + i), (as3_v4i*)(dst1 + i), 0, 0);
    };

    issue(0, 0);
    int buf = 0;
    for (int kc = 0; kc < 15; ++kc) {
        issue(kc + 1, buf ^ 1);                   // prefetch next slab
        __builtin_amdgcn_s_wait_asynccnt(8);      // oldest 8 (slab kc) landed
        __syncthreads();                          // visible to all waves
        compute(kc, buf);
        __syncthreads();                          // all done reading buf
        buf ^= 1;
    }
    __builtin_amdgcn_s_wait_asynccnt(0);
    __syncthreads();
    compute(15, buf);
#else
    // ---- fallback: synchronous stage through VGPRs --------------------------
    for (int kc = 0; kc < 16; ++kc) {
        const int r = tid * 2;
        const uint4* src = wt_u4 + (size_t)kc * 2048 + tid * 8;
        #pragma unroll
        for (int i = 0; i < 4; ++i) shB[r * 5 + i]       = src[i];
        #pragma unroll
        for (int i = 0; i < 4; ++i) shB[(r + 1) * 5 + i] = src[4 + i];
        __syncthreads();
        compute(kc, 0);
        __syncthreads();
    }
#endif

    // ---- writeback: lane (v, hi, lcol) owns out[m_base+v][col] --------------
    const int m_base = blockIdx.x * 32 + m_sub * 16 + hi * 8;
    #pragma unroll
    for (int t = 0; t < 8; ++t) {
        const int col = n_group * 128 + t * 16 + lcol;
        #pragma unroll
        for (int v = 0; v < 8; ++v) {
            out[(size_t)(m_base + v) * 512 + col] = acc[t][v];
        }
    }
}

// ---------------------------------------------------------------------------
extern "C" void kernel_launch(void* const* d_in, const int* in_sizes, int n_in,
                              void* d_out, int out_size, void* d_ws, size_t ws_size,
                              hipStream_t stream)
{
    const float* x    = (const float*)d_in[0];  // (B, 2, 256) f32
    const float* cosk = (const float*)d_in[1];  // (256, 256) f32
    const float* sink = (const float*)d_in[2];  // (256, 256) f32
    float* out        = (float*)d_out;          // (B, 512) f32
    _Float16* wtbuf   = (_Float16*)d_ws;        // 512*512 f16 = 512 KB scratch

    build_wt_kernel<<<1024, 256, 0, stream>>>(cosk, sink, wtbuf);

    const int rows = in_sizes[0] / 512;         // 65536 batches
    dft_gemm_kernel<<<rows / 32, 256, 0, stream>>>(x, wtbuf, out);
}
